// RNN_9792525435171
// MI455X (gfx1250) — matrix operand order repaired and checked by reading, not compiled, not run on hardware
//
#include <hip/hip_runtime.h>

// RNN: h_t = tanh(h_{t-1} @ Wh^T + x_t @ Wx^T + bh + bx), output h_T [B, DH]
// B=1024, T=512, D_IN=64, D_H=128.
//
// MI455X design:
//  - Mandatory HBM traffic = seq (134 MB) read once -> fully fused (no xproj).
//  - Serial in T; parallelism = 64 independent 16-row batch tiles. One wave per
//    tile, single-wave workgroups (barrier-free), 64 blocks -> 64 SIMDs busy.
//  - TRANSPOSED recurrence: h^T = tanh(Wh(A) x h^T(B) + Wx(A) x x^T(B) + b).
//    WMMA C/D layout (lane = N = batch) == B-fragment lane layout, so the
//    C->B conversion for the next step is done entirely in registers:
//    v_cvt_pk_bf16_f32 pairs + ds_swizzle_b32 SWAPX16 half-exchange.
//    -> ZERO LDS memory traffic in the 512-step serial loop.
//  - Wh/Wx A-fragments register-resident for all 512 steps (~384 VGPRs of the
//    1024-VGPR gfx1250 file). 48 v_wmma_f32_16x16x32_bf16 per step.

namespace {
constexpr int T_STEPS = 512;
constexpr int DIN     = 64;
constexpr int DH      = 128;
constexpr int ROWS_PER_WG = 16;              // one 16-row batch tile per wave
constexpr int NBLOCKS = 1024 / ROWS_PER_WG;  // 64 single-wave workgroups
}

typedef __attribute__((ext_vector_type(16))) __bf16 v16bf;
typedef __attribute__((ext_vector_type(8)))  __bf16 v8bf;
typedef __attribute__((ext_vector_type(8)))  float  v8f;

union V16 { v16bf v; v8bf h[2]; unsigned int u[8]; };

__device__ __forceinline__ float fast_tanh(float x) {
  // tanh(x) = 1 - 2/(e^{2x}+1); v_exp + v_rcp on the trans pipe (co-executes
  // with XDL WMMA). Saturates correctly at +/-1 for large |x|.
  float e = __expf(2.0f * x);
  return 1.0f - 2.0f * __builtin_amdgcn_rcpf(e + 1.0f);
}

__device__ __forceinline__ unsigned int pk_bf16(float lo, float hi) {
  union { __bf16 b[2]; unsigned int u; } t;
  t.b[0] = (__bf16)lo; t.b[1] = (__bf16)hi;   // fuses to v_cvt_pk_bf16_f32
  return t.u;
}

__device__ __forceinline__ v8bf cvt8(float4 a, float4 b) {
  v8bf r;
  r[0] = (__bf16)a.x; r[1] = (__bf16)a.y; r[2] = (__bf16)a.z; r[3] = (__bf16)a.w;
  r[4] = (__bf16)b.x; r[5] = (__bf16)b.y; r[6] = (__bf16)b.z; r[7] = (__bf16)b.w;
  return r;
}

__device__ __forceinline__ v16bf cvt16(const float* p) {
  const float4* q = (const float4*)p;
  float4 a = q[0], b = q[1], c = q[2], d = q[3];
  v16bf r;
  r[0]  = (__bf16)a.x; r[1]  = (__bf16)a.y; r[2]  = (__bf16)a.z; r[3]  = (__bf16)a.w;
  r[4]  = (__bf16)b.x; r[5]  = (__bf16)b.y; r[6]  = (__bf16)b.z; r[7]  = (__bf16)b.w;
  r[8]  = (__bf16)c.x; r[9]  = (__bf16)c.y; r[10] = (__bf16)c.z; r[11] = (__bf16)c.w;
  r[12] = (__bf16)d.x; r[13] = (__bf16)d.y; r[14] = (__bf16)d.z; r[15] = (__bf16)d.w;
  return r;
}

__global__ __launch_bounds__(32) void rnn_fused_kernel(
    const float* __restrict__ seq,   // [B, T, DIN]
    const float* __restrict__ Wh_w,  // [DH, DH]  (row n = output, col k = input)
    const float* __restrict__ Wh_b,  // [DH]
    const float* __restrict__ Wx_w,  // [DH, DIN]
    const float* __restrict__ Wx_b,  // [DH]
    float* __restrict__ out)         // [B, DH]
{
  const int lane = threadIdx.x & 31;
  const int half = lane >> 4;   // 0: lanes 0-15, 1: lanes 16-31
  const int l16  = lane & 15;
  const int rowBase = blockIdx.x * ROWS_PER_WG;

  // ---- One-time: weights as A-FRAGMENTS (M = hidden n, K = contraction dim).
  // A-frag (16x32) lane l: M = l16, K[k0..k0+15],
  // k0 = kt*32 + half*16 -> contiguous f32 in row-major weights, cvt to bf16.
  V16 whA[4][8];   // [kt][mt], 256 VGPRs
  V16 wxA[2][8];   // [kt][mt], 128 VGPRs
  float biasArr[8][8];  // [mt][v]: bias for n = mt*16 + half*8 + v
#pragma unroll
  for (int mt = 0; mt < 8; ++mt) {
    const int nA = mt * 16 + l16;        // A-frag row for this lane
#pragma unroll
    for (int kt = 0; kt < 4; ++kt) {
      const float* p = Wh_w + nA * DH + kt * 32 + half * 16;
      const float4* q = (const float4*)p;
      whA[kt][mt].h[0] = cvt8(q[0], q[1]);
      whA[kt][mt].h[1] = cvt8(q[2], q[3]);
    }
#pragma unroll
    for (int kt = 0; kt < 2; ++kt) {
      const float* p = Wx_w + nA * DIN + kt * 32 + half * 16;
      const float4* q = (const float4*)p;
      wxA[kt][mt].h[0] = cvt8(q[0], q[1]);
      wxA[kt][mt].h[1] = cvt8(q[2], q[3]);
    }
#pragma unroll
    for (int v = 0; v < 8; ++v) {
      const int n = mt * 16 + half * 8 + v;   // C-layout M index for this lane
      biasArr[mt][v] = Wx_b[n] + Wh_b[n];
    }
  }

  // h_{-1}^T = 0 : zero B-fragments (32x16 bf16 each; 4 K-tiles cover K=128)
  V16 hB[4];
#pragma unroll
  for (int kt = 0; kt < 4; ++kt) {
#pragma unroll
    for (int j = 0; j < 8; ++j) hB[kt].u[j] = 0u;
  }

  // B-fragment column for this lane is N = l16 -> batch row rowBase + l16.
  const float* mySeq = seq + (size_t)(rowBase + l16) * T_STEPS * DIN;
  float* myOut = out + (size_t)(rowBase + l16) * DH;

  for (int t = 0; t < T_STEPS; ++t) {
    // Prefetch next timestep's seq row (global_prefetch_b8).
    if (t + 1 < T_STEPS)
      __builtin_prefetch(mySeq + (t + 1) * DIN + half * 32, 0, 0);

    // --- x_t^T as B-fragments: lane l: N = l16, K = kt*32 + half*16 + 0..15
    // -> 16 contiguous f32 from this lane's seq row.
    V16 xB[2];
    const float* p = mySeq + t * DIN;
#pragma unroll
    for (int kt = 0; kt < 2; ++kt)
      xB[kt].v = cvt16(p + kt * 32 + half * 16);

    // --- 8 M-tiles: acc = bias; acc += Wx(A) x x^T(B); acc += Wh(A) x h^T(B)
    // All operands register-resident: pure WMMA, 8 independent 6-deep chains.
    unsigned int d[8][4];  // packed bf16 pairs of tanh output, [mt][pair]
#pragma unroll
    for (int mt = 0; mt < 8; ++mt) {
      v8f acc;
#pragma unroll
      for (int v = 0; v < 8; ++v) acc[v] = biasArr[mt][v];

#pragma unroll
      for (int kt = 0; kt < 2; ++kt)
        acc = __builtin_amdgcn_wmma_f32_16x16x32_bf16(
            false, wxA[kt][mt].v, false, xB[kt].v, (short)0, acc, false, false);
#pragma unroll
      for (int kt = 0; kt < 4; ++kt)
        acc = __builtin_amdgcn_wmma_f32_16x16x32_bf16(
            false, whA[kt][mt].v, false, hB[kt].v, (short)0, acc, false, false);

      float y[8];
#pragma unroll
      for (int v = 0; v < 8; ++v) y[v] = fast_tanh(acc[v]);

      // pack adjacent-M pairs: d[mt][j] holds (n_local = half*8+2j, +1), batch=l16
#pragma unroll
      for (int j = 0; j < 4; ++j) d[mt][j] = pk_bf16(y[2 * j], y[2 * j + 1]);

      // Final step: store h_T in f32 (C layout: lane's batch row, M = mt*16+half*8+v)
      if (t == T_STEPS - 1) {
        float4 s0, s1;
        s0.x = y[0]; s0.y = y[1]; s0.z = y[2]; s0.w = y[3];
        s1.x = y[4]; s1.y = y[5]; s1.z = y[6]; s1.w = y[7];
        float* o = myOut + mt * 16 + half * 8;
        *(float4*)o = s0;
        *(float4*)(o + 4) = s1;
      }
    }

    // --- C-layout -> B-fragment conversion, all in registers.
    // hB[kt] lane l: K = kt*32 + half*16 + 2v'(+1)  ->  source tile 2kt+half,
    // n_local = 2v'; n_local<8 lives in half-0 lanes, >=8 in half-1 lanes.
    // One ds_swizzle_b32 SWAPX16 (xor 16, LDS hw permute, no memory) per dword
    // exchanges the cross-half values; each direction carries a different tile.
#pragma unroll
    for (int kt = 0; kt < 4; ++kt) {
#pragma unroll
      for (int j = 0; j < 4; ++j) {
        const unsigned int loc0 = d[2 * kt][j];      // tile wanted by half-0 lanes
        const unsigned int loc1 = d[2 * kt + 1][j];  // tile wanted by half-1 lanes
        const unsigned int send = half ? loc0 : loc1;
        const unsigned int recv =
            (unsigned int)__builtin_amdgcn_ds_swizzle((int)send, 0x401f); // SWAPX16
        hB[kt].u[j]     = half ? recv : loc0;  // K pairs 2v',   v'=0..3 (n_local<8)
        hB[kt].u[4 + j] = half ? loc1 : recv;  // K pairs 2v'+8, v'=4..7 (n_local>=8)
      }
    }
  }
}

extern "C" void kernel_launch(void* const* d_in, const int* in_sizes, int n_in,
                              void* d_out, int out_size, void* d_ws, size_t ws_size,
                              hipStream_t stream) {
  const float* seq  = (const float*)d_in[0];
  const float* Wh_w = (const float*)d_in[1];
  const float* Wh_b = (const float*)d_in[2];
  const float* Wx_w = (const float*)d_in[3];
  const float* Wx_b = (const float*)d_in[4];
  float* out = (float*)d_out;

  dim3 grid(NBLOCKS);
  dim3 block(32);
  rnn_fused_kernel<<<grid, block, 0, stream>>>(seq, Wh_w, Wh_b, Wx_w, Wx_b, out);
}
